// GraphWaveNet_50878182588495
// MI455X (gfx1250) — matrix operand order repaired and checked by reading, not compiled, not run on hardware
//
#include <hip/hip_runtime.h>
#include <hip/hip_bf16.h>
#include <math.h>

// ---------------- problem constants (match reference) ----------------
#define BB 8
#define TT 32          // S
#define NN 512
#define EE 8192
#define RC 64
#define DC 64
#define SC 256
#define EC 512
#define PP 12
#define TPAD 32        // 31 valid time steps padded to 32
#define MP  (BB * NN * TPAD)   // 131072 padded GEMM rows

typedef __attribute__((ext_vector_type(16))) _Float16 v16h;
typedef __attribute__((ext_vector_type(8)))  float    v8f;

union HVec { v16h v; _Float16 h[16]; };

// ---------------- small prep kernels ----------------

__global__ void zeroK(float* A, float* deg) {
    int i = blockIdx.x * 256 + threadIdx.x;
    if (i < NN * NN) A[i] = 0.f;
    if (i < NN) deg[i] = 0.f;
}

__global__ void degK(const int* ei, const float* ew, float* deg) {
    int i = blockIdx.x * 256 + threadIdx.x;
    if (i < EE) {
        atomicAdd(&deg[ei[EE + i]], ew[i]);        // target-side degree
    } else if (i < EE + NN) {
        atomicAdd(&deg[i - EE], 1.0f);             // self loop
    }
}

__global__ void adjK(const int* ei, const float* ew, const float* deg, float* A) {
    int i = blockIdx.x * 256 + threadIdx.x;
    if (i < EE) {
        int rs = ei[i], cs = ei[EE + i];
        float dr = deg[rs] > 0.f ? rsqrtf(deg[rs]) : 0.f;
        float dc = deg[cs] > 0.f ? rsqrtf(deg[cs]) : 0.f;
        atomicAdd(&A[(size_t)cs * NN + rs], dr * ew[i] * dc);  // A[tgt, src]
    } else if (i < EE + NN) {
        int n = i - EE;
        float d = deg[n] > 0.f ? rsqrtf(deg[n]) : 0.f;
        atomicAdd(&A[(size_t)n * NN + n], d * d);
    }
}

// y[j, m] = sum_k A[m,k] * x[j,k]   (j = b*32+t), plus r[m] = rowsum(A)
__global__ void yK(const float* A, const float* x, float* y, float* r) {
    int m = blockIdx.x * 256 + threadIdx.x;     // 0..511 (grid.x = 2)
    int j = blockIdx.y;                         // 0..256 (256 = rowsum job)
    const float* Ar = A + (size_t)m * NN;
    float acc = 0.f;
    if (j < BB * TT) {
        const float* xr = x + (size_t)j * NN;
        for (int k = 0; k < NN; ++k) acc += Ar[k] * xr[k];
        y[(size_t)j * NN + m] = acc;
    } else {
        for (int k = 0; k < NN; ++k) acc += Ar[k];
        r[m] = acc;
    }
}

// Collapse start->filt/gate->gcn_w into per-channel coefficient vectors.
// coeff layout: [a0g | a1g | cg | w0 | w1 | wc], 64 floats each.
__global__ void coefK(const float* sw, const float* sb0, const float* fw, const float* fb,
                      const float* gw, const float* gb, const float* gcw, float* coeff) {
    __shared__ float sa0[DC], sa1[DC], scf[DC];
    int o = threadIdx.x;           // 0..63
    float a0f = 0, a1f = 0, cf = 0, a0g = 0, a1g = 0, cg = 0;
    for (int c = 0; c < RC; ++c) {
        float s = sw[c], b = sb0[c];
        float f0 = fw[(o * RC + c) * 2 + 0], f1 = fw[(o * RC + c) * 2 + 1];
        float g0 = gw[(o * RC + c) * 2 + 0], g1 = gw[(o * RC + c) * 2 + 1];
        a0f += f0 * s; a1f += f1 * s; cf += (f0 + f1) * b;
        a0g += g0 * s; a1g += g1 * s; cg += (g0 + g1) * b;
    }
    cf += fb[o]; cg += gb[o];
    sa0[o] = a0f; sa1[o] = a1f; scf[o] = cf;
    coeff[o] = a0g; coeff[64 + o] = a1g; coeff[128 + o] = cg;
    __syncthreads();
    float w0 = 0, w1 = 0, wc = 0;
    for (int c = 0; c < DC; ++c) {
        float g = gcw[o * DC + c];
        w0 += g * sa0[c]; w1 += g * sa1[c]; wc += g * scf[c];
    }
    coeff[192 + o] = w0; coeff[256 + o] = w1; coeff[320 + o] = wc;
}

// f32 -> f16 weight copies for the WMMA GEMMs
__global__ void cvtK(const float* skw, const float* e1w, _Float16* wsk, _Float16* we1) {
    int i = blockIdx.x * 256 + threadIdx.x;
    if (i < SC * DC) wsk[i] = (_Float16)skw[i];
    else if (i < SC * DC + EC * SC) we1[i - SC * DC] = (_Float16)e1w[i - SC * DC];
}

// hg[m, c] = tanh(fg) * sigmoid(g), row m = (b*512+n)*32 + t, padded t==31 row -> 0
__global__ void hgK(const float* x, const float* y, const float* r,
                    const float* coeff, const float* gcb, _Float16* Hg) {
    int m = blockIdx.x * 256 + threadIdx.x;    // 0..MP-1
    int t = m & 31;
    int site = m >> 5;
    int n = site & (NN - 1);
    int b = site >> 9;
    _Float16* row = Hg + (size_t)m * DC;
    if (t == TPAD - 1) {
        for (int c = 0; c < DC; ++c) row[c] = (_Float16)0.f;
        return;
    }
    float xv0 = x[((size_t)(b * TT + t)) * NN + n];
    float xv1 = x[((size_t)(b * TT + t + 1)) * NN + n];
    float yv0 = y[((size_t)(b * TT + t)) * NN + n];
    float yv1 = y[((size_t)(b * TT + t + 1)) * NN + n];
    float rv = r[n];
    for (int c = 0; c < DC; ++c) {
        float fg = coeff[192 + c] * yv0 + coeff[256 + c] * yv1 + coeff[320 + c] * rv + gcb[c];
        float gv = coeff[c] * xv0 + coeff[64 + c] * xv1 + coeff[128 + c];
        float hv = tanhf(fg) * (1.f / (1.f + __expf(-gv)));
        row[c] = (_Float16)hv;
    }
}

// ---------------- WMMA GEMM 1: Z1[MP,256] = relu(Hg[MP,64] @ skip_w^T + skip_b) ----------------
// One wave owns a 16-row strip; A fragments loaded once and reused across 4 channel tiles.
__global__ void gemm1K(const _Float16* __restrict__ Hg, const _Float16* __restrict__ Wsk,
                       const float* __restrict__ skb, _Float16* __restrict__ Z1) {
    int lane = threadIdx.x & 31, wave = threadIdx.x >> 5;
    int l = lane & 15, h = lane >> 4;
    int row0 = blockIdx.x * 16;
    HVec a[2];
#pragma unroll
    for (int kk = 0; kk < 2; ++kk) {
        const _Float16* arow = Hg + (size_t)(row0 + l) * DC + kk * 32;
#pragma unroll
        for (int e = 0; e < 16; ++e) {
            int kl = h * 16 + e;   // A: lane-half = K-half (contiguous, matches B)
            a[kk].h[e] = arow[kl];
        }
    }
#pragma unroll
    for (int i = 0; i < 4; ++i) {
        int ct = (wave * 4 + i) * 16;
        v8f acc = {};
#pragma unroll
        for (int kk = 0; kk < 2; ++kk) {
            HVec bm;
            const _Float16* brow = Wsk + (size_t)(ct + l) * DC + kk * 32 + h * 16;
#pragma unroll
            for (int e = 0; e < 16; ++e) bm.h[e] = brow[e];          // B: lane-half = K-half
            acc = __builtin_amdgcn_wmma_f32_16x16x32_f16(false, a[kk].v, false, bm.v,
                                                         (short)0, acc, false, false);
        }
        float bias = skb[ct + l];
#pragma unroll
        for (int rr = 0; rr < 8; ++rr) {
            float vv = acc[rr] + bias;
            vv = vv > 0.f ? vv : 0.f;
            Z1[(size_t)(row0 + h * 8 + rr) * SC + ct + l] = (_Float16)vv;
        }
    }
}

// ---- WMMA GEMM 2 + fused time-mean: zbar[site,512] = sum_{t<31} relu(Z1 @ end1_w^T + end1_b) ----
// One wave holds full-K A fragments (8 x v16h) and sweeps 4 channel tiles; no atomics.
__global__ void gemm2K(const _Float16* __restrict__ Z1, const _Float16* __restrict__ We1,
                       const float* __restrict__ e1b, float* __restrict__ zbar) {
    int lane = threadIdx.x & 31, wave = threadIdx.x >> 5;
    int l = lane & 15, h = lane >> 4;
    int site = blockIdx.x;
    int g = blockIdx.y * 4 + wave;                 // 0..7 channel-tile group
    if (site + 1 < BB * NN)                        // warm L2 for the next site's A rows
        __builtin_prefetch(Z1 + ((size_t)(site + 1) * TPAD + l) * SC, 0, 0);
    float colsum[4] = {0.f, 0.f, 0.f, 0.f};
    for (int mt = 0; mt < 2; ++mt) {
        int row0 = site * TPAD + mt * 16;
        HVec a[8];
#pragma unroll
        for (int kk = 0; kk < 8; ++kk) {
            const _Float16* arow = Z1 + (size_t)(row0 + l) * SC + kk * 32;
#pragma unroll
            for (int e = 0; e < 16; ++e) {
                int kl = h * 16 + e;
                a[kk].h[e] = arow[kl];
            }
        }
#pragma unroll
        for (int i = 0; i < 4; ++i) {
            int ch = (g * 4 + i) * 16 + l;
            v8f acc = {};
#pragma unroll
            for (int kk = 0; kk < 8; ++kk) {
                HVec bm;
                const _Float16* brow = We1 + (size_t)ch * SC + kk * 32 + h * 16;
#pragma unroll
                for (int e = 0; e < 16; ++e) bm.h[e] = brow[e];
                acc = __builtin_amdgcn_wmma_f32_16x16x32_f16(false, a[kk].v, false, bm.v,
                                                             (short)0, acc, false, false);
            }
            float bias = e1b[ch];
#pragma unroll
            for (int rr = 0; rr < 8; ++rr) {
                int t = mt * 16 + h * 8 + rr;
                float vv = acc[rr] + bias;
                vv = vv > 0.f ? vv : 0.f;
                if (t < TPAD - 1) colsum[i] += vv;     // drop padded t==31 row
            }
        }
    }
#pragma unroll
    for (int i = 0; i < 4; ++i) {
        float c = colsum[i] + __shfl_xor(colsum[i], 16, 32);   // combine 8-row halves (wave32)
        if (h == 0) zbar[(size_t)site * EC + (g * 4 + i) * 16 + l] = c;
    }
}

// out[b,p,n] = end2_w @ (zbar/31) + end2_b
__global__ void outK(const float* zbar, const float* w2, const float* b2, float* out) {
    int site = blockIdx.x;
    int p = threadIdx.x;
    if (p >= PP) return;
    int b = site >> 9, n = site & (NN - 1);
    const float* zr = zbar + (size_t)site * EC;
    float acc = 0.f;
    for (int k = 0; k < EC; ++k) acc += w2[p * EC + k] * zr[k];
    out[((size_t)(b * PP + p)) * NN + n] = acc * (1.f / 31.f) + b2[p];
}

// ---------------- launcher ----------------
extern "C" void kernel_launch(void* const* d_in, const int* in_sizes, int n_in,
                              void* d_out, int out_size, void* d_ws, size_t ws_size,
                              hipStream_t stream) {
    const float* x    = (const float*)d_in[0];
    const int*   ei   = (const int*)d_in[1];
    const float* ew   = (const float*)d_in[2];
    const float* sw   = (const float*)d_in[3];
    const float* sb0  = (const float*)d_in[4];
    const float* fw   = (const float*)d_in[5];
    const float* fb   = (const float*)d_in[6];
    const float* gw   = (const float*)d_in[7];
    const float* gb   = (const float*)d_in[8];
    const float* gcw  = (const float*)d_in[9];
    const float* gcb  = (const float*)d_in[10];
    // d_in[11], d_in[12] (res_w/res_b): dead code in reference output
    const float* skw  = (const float*)d_in[13];
    const float* skb  = (const float*)d_in[14];
    const float* e1w  = (const float*)d_in[15];
    const float* e1b  = (const float*)d_in[16];
    const float* w2   = (const float*)d_in[17];
    const float* b2   = (const float*)d_in[18];
    float* out = (float*)d_out;

    char* ws = (char*)d_ws;
    size_t off = 0;
    float*    A     = (float*)(ws + off); off += (size_t)NN * NN * 4;              // 1 MB
    float*    deg   = (float*)(ws + off); off += NN * 4ull;                        // 2 KB
    float*    y     = (float*)(ws + off); off += (size_t)BB * TT * NN * 4;         // 512 KB
    float*    rsum  = (float*)(ws + off); off += NN * 4ull;
    float*    coeff = (float*)(ws + off); off += 6 * 64 * 4ull; off = (off + 255) & ~255ull;
    _Float16* wsk   = (_Float16*)(ws + off); off += (size_t)SC * DC * 2;           // 32 KB
    _Float16* we1   = (_Float16*)(ws + off); off += (size_t)EC * SC * 2;           // 256 KB
    _Float16* Hg    = (_Float16*)(ws + off); off += (size_t)MP * DC * 2;           // 16 MB
    _Float16* Z1    = (_Float16*)(ws + off); off += (size_t)MP * SC * 2;           // 64 MB
    float*    zbar  = (float*)(ws + off); off += (size_t)BB * NN * EC * 4;         // 8 MB

    // 1. zero adjacency + degree
    zeroK<<<dim3((NN * NN + 255) / 256), 256, 0, stream>>>(A, deg);
    // 2. degree accumulation (edges + self loops)
    degK<<<dim3((EE + NN + 255) / 256), 256, 0, stream>>>(ei, ew, deg);
    // 3. scatter normalized adjacency A[tgt,src]
    adjK<<<dim3((EE + NN + 255) / 256), 256, 0, stream>>>(ei, ew, deg, A);
    // 4. y = A @ x per (b,t); r = rowsum(A)
    yK<<<dim3(NN / 256, BB * TT + 1), 256, 0, stream>>>(A, x, y, rsum);
    // 5. coefficient vectors (rank-1 channel collapse)
    coefK<<<1, 64, 0, stream>>>(sw, sb0, fw, fb, gw, gb, gcw, coeff);
    // 6. f16 weight conversion
    cvtK<<<dim3((SC * DC + EC * SC + 255) / 256), 256, 0, stream>>>(skw, e1w, wsk, we1);
    // 7. gated activation hg -> f16 [MP,64]
    hgK<<<dim3(MP / 256), 256, 0, stream>>>(x, y, rsum, coeff, gcb, Hg);
    // 8. WMMA GEMM: skip + relu -> Z1 f16 [MP,256], 8 wmma/wave, A reused in regs
    gemm1K<<<dim3(MP / 16), 128, 0, stream>>>(Hg, wsk, skb, Z1);
    // 9. WMMA GEMM: end1 + relu + fused time-sum -> zbar [B*N,512], 64 wmma/wave
    gemm2K<<<dim3(BB * NN, 2), 128, 0, stream>>>(Z1, we1, e1b, zbar);
    // 10. end2 matvec + mean scale -> out [B,P,N]
    outK<<<dim3(BB * NN), 32, 0, stream>>>(zbar, w2, b2, out);
}
